// LocalDecoder_38809324487321
// MI455X (gfx1250) — matrix operand order repaired and checked by reading, not compiled, not run on hardware
//
#include <hip/hip_runtime.h>
#include <hip/hip_bf16.h>

// ---------------- problem constants (match reference) ----------------
constexpr int Bc = 4, TYc = 128, TXc = 512, Dc = 256, DFFc = 1024, Hc = 8, NLc = 6;
constexpr int VOCABc = 32000, MAXEXTc = 200, DKc = 32;
constexpr int BT = Bc * TYc;   // 512 decoder rows
constexpr int BX = Bc * TXc;   // 2048 memory rows
constexpr int OUTW = VOCABc + MAXEXTc; // 32200
constexpr float EPSc = 1e-6f;

typedef __attribute__((ext_vector_type(16))) __bf16 v16bf;
typedef __attribute__((ext_vector_type(8)))  float  v8f;

// ---------------- small helpers ----------------
__device__ inline float warp_sum(float v) {
  #pragma unroll
  for (int o = 16; o; o >>= 1) v += __shfl_xor(v, o, 32);
  return v;
}
__device__ inline float warp_max(float v) {
  #pragma unroll
  for (int o = 16; o; o >>= 1) v = fmaxf(v, __shfl_xor(v, o, 32));
  return v;
}
__device__ inline float block_sum(float v, float* red) {
  int lane = threadIdx.x & 31, w = threadIdx.x >> 5, nw = (blockDim.x + 31) >> 5;
  v = warp_sum(v);
  if (lane == 0) red[w] = v;
  __syncthreads();
  if (w == 0) {
    float x = (lane < nw) ? red[lane] : 0.f;
    x = warp_sum(x);
    if (lane == 0) red[0] = x;
  }
  __syncthreads();
  float r = red[0];
  __syncthreads();
  return r;
}
__device__ inline float block_max(float v, float* red) {
  int lane = threadIdx.x & 31, w = threadIdx.x >> 5, nw = (blockDim.x + 31) >> 5;
  v = warp_max(v);
  if (lane == 0) red[w] = v;
  __syncthreads();
  if (w == 0) {
    float x = (lane < nw) ? red[lane] : -3.4e38f;
    x = warp_max(x);
    if (lane == 0) red[0] = x;
  }
  __syncthreads();
  float r = red[0];
  __syncthreads();
  return r;
}

// ---------------- fp32 -> bf16 weight conversion ----------------
__global__ __launch_bounds__(256) void cvt_bf16_kernel(const float* __restrict__ src,
                                                       __bf16* __restrict__ dst, int n) {
  int i = blockIdx.x * 256 + threadIdx.x;
  if (i < n) dst[i] = (__bf16)src[i];
}

// ---------------- embedding ----------------
__global__ __launch_bounds__(256) void embed_kernel(const int* __restrict__ x,
                                                    const int* __restrict__ p,
                                                    const float* __restrict__ w_emb,
                                                    const float* __restrict__ p_emb,
                                                    float* __restrict__ emb,
                                                    float* __restrict__ h) {
  int row = blockIdx.x;              // 0..BT-1
  int d   = threadIdx.x;             // 0..D-1
  int xi = x[row], pi = p[row];
  float e = w_emb[(size_t)xi * Dc + d] + p_emb[(size_t)pi * Dc + d];
  emb[(size_t)row * Dc + d] = e;
  h[(size_t)row * Dc + d]   = e;
}

// ---------------- layernorm (ddof = 1, like reference) ----------------
__global__ __launch_bounds__(256) void layernorm_kernel(const float* __restrict__ x,
                                                        const float* __restrict__ a,
                                                        const float* __restrict__ bsh,
                                                        float* __restrict__ out) {
  __shared__ float red[32];
  int row = blockIdx.x, d = threadIdx.x;
  float v = x[(size_t)row * Dc + d];
  float mu = block_sum(v, red) * (1.f / Dc);
  float dv = v - mu;
  float var = block_sum(dv * dv, red) * (1.f / (Dc - 1));
  float sd = sqrtf(var);
  out[(size_t)row * Dc + d] = a[d] * dv / (sd + EPSc) + bsh[d];
}

// ---------------- WMMA bf16 GEMM:  C = act(A @ W^T + bias) [+ res] ----------------
// A: fp32 [M,K] (converted to bf16 on the fly), W: bf16 [N,K], C: fp32 row stride ldC.
// Double-buffered pipeline:
//   B tiles: CDNA5 async global->LDS DMA (ASYNCcnt), tile i+1 in flight while
//            tile i computes; "s_wait_asynccnt 2" retires tile i only (in-order).
//   A tiles: global->VGPR prefetch one iteration ahead, cvt+ds_store to alt buffer.
// Requires M%64==0, N%128==0, K%32==0 (all call sites satisfy this).
__global__ __launch_bounds__(256)
void gemm_bf16_wmma_kernel(const float* __restrict__ A, const __bf16* __restrict__ W,
                           const float* __restrict__ bias, const float* __restrict__ res,
                           float* __restrict__ C, int M, int N, int K, int ldC, int doRelu) {
  __shared__ __align__(16) __bf16 As[2][64][32];
  __shared__ __align__(16) __bf16 Bs[2][128][32];
  const int tid  = threadIdx.x;
  const int lane = tid & 31, wid = tid >> 5;
  const int wm = wid >> 2, wn = wid & 3;           // 2 x 4 wave grid over the block tile
  const int bm = blockIdx.y * 64, bn = blockIdx.x * 128;
  const int half = lane >> 4, l16 = lane & 15;

  // per-thread staging coordinates
  const int br = tid >> 1, bkc = (tid & 1) << 4;   // B: row, 16 consecutive k (2 x b128)
  const int ar = tid >> 2, akc = (tid & 3) << 3;   // A: row, 8 consecutive k
  const unsigned blds[2] = {
    (unsigned)(unsigned long long)(const void*)&Bs[0][br][bkc],
    (unsigned)(unsigned long long)(const void*)&Bs[1][br][bkc] };

  auto issueB = [&](int kstep, int buf) {
    const __bf16* gsrc = W + (size_t)(bn + br) * K + (kstep << 5) + bkc;
    asm volatile("global_load_async_to_lds_b128 %0, %1, off"
                 :: "v"(blds[buf]), "v"(gsrc) : "memory");
    asm volatile("global_load_async_to_lds_b128 %0, %1, off"
                 :: "v"(blds[buf] + 16u), "v"(gsrc + 8) : "memory");
  };
  auto loadA = [&](int kstep, float4& f0, float4& f1) {
    const float* src = A + (size_t)(bm + ar) * K + (kstep << 5) + akc;
    f0 = *(const float4*)(src);
    f1 = *(const float4*)(src + 4);
  };
  auto storeA = [&](int buf, const float4& f0, const float4& f1) {
    __bf16* dst = &As[buf][ar][akc];
    dst[0] = (__bf16)f0.x; dst[1] = (__bf16)f0.y; dst[2] = (__bf16)f0.z; dst[3] = (__bf16)f0.w;
    dst[4] = (__bf16)f1.x; dst[5] = (__bf16)f1.y; dst[6] = (__bf16)f1.z; dst[7] = (__bf16)f1.w;
  };

  v8f acc[2][2] = {};
  union Frag { v16bf v; unsigned int u[8]; };

  const int nk = K >> 5;
  float4 a0, a1, p0, p1;
  issueB(0, 0);
  loadA(0, a0, a1);

  for (int i = 0; i < nk; ++i) {
    const int cur = i & 1, nxt = cur ^ 1;
    const bool more = (i + 1 < nk);
    if (more) issueB(i + 1, nxt);           // DMA next B tile while tile i computes
    storeA(cur, a0, a1);                    // cvt + ds_store this A tile
    if (more) loadA(i + 1, p0, p1);         // prefetch next A tile into VGPRs
    if (more) asm volatile("s_wait_asynccnt 2" ::: "memory");  // tile i landed
    else      asm volatile("s_wait_asynccnt 0" ::: "memory");
    __syncthreads();

    // build fragments per ISA 7.12.2 VGPR layouts (16-bit A 16x32 / B 32x16)
    Frag af[2], bf[2];
    #pragma unroll
    for (int tm = 0; tm < 2; ++tm) {
      int r = wm * 32 + tm * 16 + l16;
      #pragma unroll
      for (int p = 0; p < 8; ++p) {
        int kk = (p < 4 ? 2 * p : 16 + 2 * (p - 4)) + 8 * half;   // even
        af[tm].u[p] = *(const unsigned int*)&As[cur][r][kk];
      }
    }
    #pragma unroll
    for (int tn = 0; tn < 2; ++tn) {
      int c = wn * 32 + tn * 16 + l16;
      #pragma unroll
      for (int p = 0; p < 8; ++p) {
        int kk = (p < 4 ? 2 * p : 16 + 2 * (p - 4)) + 8 * half;   // even
        bf[tn].u[p] = *(const unsigned int*)&Bs[cur][c][kk];
      }
    }
    acc[0][0] = __builtin_amdgcn_wmma_f32_16x16x32_bf16(false, af[0].v, false, bf[0].v,
                                                        (short)0, acc[0][0], false, false);
    acc[0][1] = __builtin_amdgcn_wmma_f32_16x16x32_bf16(false, af[0].v, false, bf[1].v,
                                                        (short)0, acc[0][1], false, false);
    acc[1][0] = __builtin_amdgcn_wmma_f32_16x16x32_bf16(false, af[1].v, false, bf[0].v,
                                                        (short)0, acc[1][0], false, false);
    acc[1][1] = __builtin_amdgcn_wmma_f32_16x16x32_bf16(false, af[1].v, false, bf[1].v,
                                                        (short)0, acc[1][1], false, false);
    a0 = p0; a1 = p1;
    __syncthreads();   // reads of buffer `cur` done before it is overwritten
  }

  // epilogue: C/D 16x16 layout -> lane gives N=l16, VGPR r gives M=r+8*half
  #pragma unroll
  for (int tm = 0; tm < 2; ++tm) {
    #pragma unroll
    for (int tn = 0; tn < 2; ++tn) {
      int col = bn + wn * 32 + tn * 16 + l16;
      float bvv = bias ? bias[col] : 0.f;
      #pragma unroll
      for (int r = 0; r < 8; ++r) {
        int row = bm + wm * 32 + tm * 16 + half * 8 + r;
        float v = acc[tm][tn][r] + bvv;
        if (doRelu) v = fmaxf(v, 0.f);
        if (res) v += res[(size_t)row * N + col];
        C[(size_t)row * ldC + col] = v;
      }
    }
  }
}

// ---------------- fused softmax attention (wave per (b,h,q-row)) ----------------
// q: [BT, D] (head h at cols h*32..), k,v: [B*Tk, D]; out: [BT, D]
__global__ __launch_bounds__(256)
void attention_kernel(const float* __restrict__ q, const float* __restrict__ k,
                      const float* __restrict__ v, const float* __restrict__ mask,
                      int Tk, int selfMode, float* __restrict__ out) {
  __shared__ float qsh[8][DKc];
  __shared__ float sc[8][TXc];
  const int lane = threadIdx.x & 31, wid = threadIdx.x >> 5;
  const int idx = blockIdx.x * 8 + wid;          // (b,h,qr)
  const int b  = idx / (Hc * TYc);
  const int hh = (idx / TYc) % Hc;
  const int qr = idx % TYc;
  const float scale = 0.17677669529663687f;      // 1/sqrt(32)

  qsh[wid][lane] = q[((size_t)(b * TYc + qr)) * Dc + hh * DKc + lane] * scale;
  __syncthreads();

  float lmax = -3.4e38f;
  for (int kc = lane; kc < Tk; kc += 32) {
    const float* kp = k + ((size_t)(b * Tk + kc)) * Dc + hh * DKc;
    float s = 0.f;
    #pragma unroll
    for (int d = 0; d < DKc; ++d) s += qsh[wid][d] * kp[d];
    float mval = selfMode ? mask[((size_t)(b * TYc + qr)) * TYc + kc]
                          : mask[(size_t)b * TXc + kc];
    if (mval == 0.f) s = -1e9f;
    sc[wid][kc] = s;
    lmax = fmaxf(lmax, s);
  }
  lmax = warp_max(lmax);
  float lsum = 0.f;
  for (int kc = lane; kc < Tk; kc += 32) {
    float e = __expf(sc[wid][kc] - lmax);
    sc[wid][kc] = e;
    lsum += e;
  }
  lsum = warp_sum(lsum);
  float inv = 1.f / lsum;

  // o[dk] pass: one lane per channel (DK == 32 == wave size)
  float o = 0.f;
  for (int kc = 0; kc < Tk; ++kc)
    o += sc[wid][kc] * v[((size_t)(b * Tk + kc)) * Dc + hh * DKc + lane];
  out[((size_t)(b * TYc + qr)) * Dc + hh * DKc + lane] = o * inv;
}

// ---------------- vocab softmax in-place on d_out rows; zero ext region ----------------
__global__ __launch_bounds__(256) void softmax_vocab_kernel(float* __restrict__ out) {
  __shared__ float red[32];
  int row = blockIdx.x;
  float* p = out + (size_t)row * OUTW;
  float m = -3.4e38f;
  for (int j = threadIdx.x; j < VOCABc; j += 256) m = fmaxf(m, p[j]);
  m = block_max(m, red);
  float s = 0.f;
  for (int j = threadIdx.x; j < VOCABc; j += 256) s += __expf(p[j] - m);
  s = block_sum(s, red);
  float inv = 1.f / s;
  for (int j = threadIdx.x; j < VOCABc; j += 256) p[j] = __expf(p[j] - m) * inv;
  for (int j = VOCABc + threadIdx.x; j < OUTW; j += 256) p[j] = 0.f;
}

// ---------------- fused copy attention: scores -> softmax(w) -> context -> gate ----------------
// block per (b,y). Writes w[b,x,y] (for scatter) and g[b,y].
__global__ __launch_bounds__(256)
void copy_attn_kernel(const float* __restrict__ pctx, const float* __restrict__ hW,
                      const float* __restrict__ U, const float* __restrict__ mask_x,
                      const float* __restrict__ memory, const float* __restrict__ emb,
                      const float* __restrict__ hf, const float* __restrict__ vvec,
                      const float* __restrict__ bv, float* __restrict__ wout,
                      float* __restrict__ gout) {
  __shared__ float hWs[Dc];
  __shared__ float sc[TXc];
  __shared__ float red[32];
  const int tid = threadIdx.x, lane = tid & 31, wid = tid >> 5;
  const int b = blockIdx.x / TYc, yy = blockIdx.x % TYc;

  hWs[tid] = hW[((size_t)(b * TYc + yy)) * Dc + tid];
  __syncthreads();

  for (int x = wid; x < TXc; x += 8) {
    const float* pr = pctx + ((size_t)(b * TXc + x)) * Dc;
    float s = 0.f;
    #pragma unroll
    for (int d = lane; d < Dc; d += 32) s += tanhf(pr[d] + hWs[d]) * U[d];
    s = warp_sum(s);
    if (lane == 0) sc[x] = s * mask_x[(size_t)b * TXc + x];
  }
  __syncthreads();

  // softmax over x (two entries per thread)
  float m = fmaxf(sc[tid], sc[tid + 256]);
  m = block_max(m, red);
  float e0 = __expf(sc[tid] - m)       * mask_x[(size_t)b * TXc + tid];
  float e1 = __expf(sc[tid + 256] - m) * mask_x[(size_t)b * TXc + tid + 256];
  float ssum = block_sum(e0 + e1, red);
  float inv = 1.f / ssum;
  sc[tid] = e0 * inv; sc[tid + 256] = e1 * inv;
  wout[((size_t)(b * TXc + tid)) * TYc + yy]       = sc[tid];
  wout[((size_t)(b * TXc + tid + 256)) * TYc + yy] = sc[tid + 256];
  __syncthreads();

  // context atts[d] for d = tid, then gate partial
  float a = 0.f;
  for (int x = 0; x < TXc; ++x) a += sc[x] * memory[((size_t)(b * TXc + x)) * Dc + tid];
  size_t ro = ((size_t)(b * TYc + yy)) * Dc;
  float gp = emb[ro + tid] * vvec[tid] + hf[ro + tid] * vvec[Dc + tid] + a * vvec[2 * Dc + tid];
  float gs = block_sum(gp, red);
  if (tid == 0) gout[b * TYc + yy] = 1.f / (1.f + __expf(-(gs + bv[0])));
}

// ---------------- gate scale + pointer scatter-add ----------------
__global__ __launch_bounds__(256)
void combine_kernel(float* __restrict__ out, const float* __restrict__ g,
                    const float* __restrict__ wbuf, const int* __restrict__ xid) {
  const int row = blockIdx.x;
  const int b = row / TYc, yy = row % TYc;
  float gg = g[row];
  float* p = out + (size_t)row * OUTW;
  for (int j = threadIdx.x; j < VOCABc; j += 256) p[j] *= gg;
  __syncthreads();
  float og = 1.f - gg;
  for (int x = threadIdx.x; x < TXc; x += 256) {
    int id = xid[b * TXc + x];
    atomicAdd(&p[id], og * wbuf[((size_t)(b * TXc + x)) * TYc + yy]);
  }
}

// ---------------- host orchestration ----------------
extern "C" void kernel_launch(void* const* d_in, const int* in_sizes, int n_in,
                              void* d_out, int out_size, void* d_ws, size_t ws_size,
                              hipStream_t stream) {
  const int*   x_     = (const int*)d_in[0];
  const int*   p_     = (const int*)d_in[1];
  const float* memory = (const float*)d_in[3];
  const float* mask_x = (const float*)d_in[4];
  const float* mask_y = (const float*)d_in[5];
  const int*   xid    = (const int*)d_in[6];
  const float* w_emb  = (const float*)d_in[8];
  const float* p_emb  = (const float*)d_in[9];
  const float* Wqkvs  = (const float*)d_in[10];
  const float* bqkvs  = (const float*)d_in[11];
  const float* Wos    = (const float*)d_in[12];
  const float* bos    = (const float*)d_in[13];
  const float* Wqkvx  = (const float*)d_in[14];
  const float* bqkvx  = (const float*)d_in[15];
  const float* Wox    = (const float*)d_in[16];
  const float* box    = (const float*)d_in[17];
  const float* W1     = (const float*)d_in[18];
  const float* b1     = (const float*)d_in[19];
  const float* W2     = (const float*)d_in[20];
  const float* b2     = (const float*)d_in[21];
  const float* ln_s   = (const float*)d_in[22];
  const float* ln_b   = (const float*)d_in[23];
  const float* fln_s  = (const float*)d_in[24];
  const float* fln_b  = (const float*)d_in[25];
  const float* projW  = (const float*)d_in[26];
  const float* projb  = (const float*)d_in[27];
  const float* WcAtt  = (const float*)d_in[28];
  const float* bAtt   = (const float*)d_in[29];
  const float* Wcomb  = (const float*)d_in[30];
  const float* Uatt   = (const float*)d_in[31];
  const float* vvec   = (const float*)d_in[32];
  const float* bv     = (const float*)d_in[33];
  float* out = (float*)d_out;

  // ---- workspace layout ----
  size_t off = 0;
  auto allocF = [&](size_t n) -> float* {
    float* r = (float*)((char*)d_ws + off);
    off = (off + n * sizeof(float) + 255) & ~(size_t)255;
    return r;
  };
  auto allocB = [&](size_t n) -> __bf16* {
    __bf16* r = (__bf16*)((char*)d_ws + off);
    off = (off + n * sizeof(__bf16) + 255) & ~(size_t)255;
    return r;
  };
  float* emb  = allocF((size_t)BT * Dc);
  float* h    = allocF((size_t)BT * Dc);
  float* y    = allocF((size_t)BT * Dc);
  float* qb   = allocF((size_t)BT * Dc);
  float* kb   = allocF((size_t)BX * Dc);
  float* vb   = allocF((size_t)BX * Dc);
  float* ao   = allocF((size_t)BT * Dc);
  float* ff   = allocF((size_t)BT * DFFc);
  float* wbuf = allocF((size_t)BX * TYc);
  float* pctx = allocF((size_t)BX * Dc);
  float* hWb  = allocF((size_t)BT * Dc);
  float* gbuf = allocF((size_t)BT);
  __bf16* cQKVs = allocB((size_t)NLc * 3 * Dc * Dc);
  __bf16* cOs   = allocB((size_t)NLc * Dc * Dc);
  __bf16* cQKVx = allocB((size_t)NLc * 3 * Dc * Dc);
  __bf16* cOx   = allocB((size_t)NLc * Dc * Dc);
  __bf16* cW1   = allocB((size_t)NLc * DFFc * Dc);
  __bf16* cW2   = allocB((size_t)NLc * Dc * DFFc);
  __bf16* cProj = allocB((size_t)VOCABc * Dc);
  __bf16* cC    = allocB((size_t)Dc * Dc);
  __bf16* cComb = allocB((size_t)Dc * Dc);

  auto cvt = [&](const float* s, __bf16* d, size_t n) {
    cvt_bf16_kernel<<<dim3((unsigned)((n + 255) / 256)), dim3(256), 0, stream>>>(s, d, (int)n);
  };
  cvt(Wqkvs, cQKVs, (size_t)NLc * 3 * Dc * Dc);
  cvt(Wos,   cOs,   (size_t)NLc * Dc * Dc);
  cvt(Wqkvx, cQKVx, (size_t)NLc * 3 * Dc * Dc);
  cvt(Wox,   cOx,   (size_t)NLc * Dc * Dc);
  cvt(W1,    cW1,   (size_t)NLc * DFFc * Dc);
  cvt(W2,    cW2,   (size_t)NLc * Dc * DFFc);
  cvt(projW, cProj, (size_t)VOCABc * Dc);
  cvt(WcAtt, cC,    (size_t)Dc * Dc);
  cvt(Wcomb, cComb, (size_t)Dc * Dc);

  auto gemm = [&](const float* A, const __bf16* W, const float* bias, const float* res,
                  float* C, int M, int N, int K, int ldC, int relu) {
    dim3 grid(N / 128, M / 64);
    gemm_bf16_wmma_kernel<<<grid, dim3(256), 0, stream>>>(A, W, bias, res, C, M, N, K, ldC, relu);
  };
  auto lnorm = [&](const float* xin, const float* a, const float* bb, float* o) {
    layernorm_kernel<<<dim3(BT), dim3(Dc), 0, stream>>>(xin, a, bb, o);
  };

  embed_kernel<<<dim3(BT), dim3(Dc), 0, stream>>>(x_, p_, w_emb, p_emb, emb, h);

  const int attnBlocks = (Bc * Hc * TYc) / 8; // wave per (b,h,q-row)
  for (int i = 0; i < NLc; ++i) {
    // --- self attention ---
    lnorm(h, ln_s + (size_t)(i * 3 + 0) * Dc, ln_b + (size_t)(i * 3 + 0) * Dc, y);
    for (int j = 0; j < 3; ++j) {
      float* dst = (j == 0) ? qb : (j == 1) ? kb : vb;
      gemm(y, cQKVs + (size_t)(i * 3 + j) * Dc * Dc, bqkvs + (size_t)(i * 3 + j) * Dc,
           nullptr, dst, BT, Dc, Dc, Dc, 0);
    }
    attention_kernel<<<dim3(attnBlocks), dim3(256), 0, stream>>>(qb, kb, vb, mask_y, TYc, 1, ao);
    gemm(ao, cOs + (size_t)i * Dc * Dc, bos + (size_t)i * Dc, h, h, BT, Dc, Dc, Dc, 0);

    // --- cross attention ---
    lnorm(h, ln_s + (size_t)(i * 3 + 1) * Dc, ln_b + (size_t)(i * 3 + 1) * Dc, y);
    gemm(y,      cQKVx + (size_t)(i * 3 + 0) * Dc * Dc, bqkvx + (size_t)(i * 3 + 0) * Dc,
         nullptr, qb, BT, Dc, Dc, Dc, 0);
    gemm(memory, cQKVx + (size_t)(i * 3 + 1) * Dc * Dc, bqkvx + (size_t)(i * 3 + 1) * Dc,
         nullptr, kb, BX, Dc, Dc, Dc, 0);
    gemm(memory, cQKVx + (size_t)(i * 3 + 2) * Dc * Dc, bqkvx + (size_t)(i * 3 + 2) * Dc,
         nullptr, vb, BX, Dc, Dc, Dc, 0);
    attention_kernel<<<dim3(attnBlocks), dim3(256), 0, stream>>>(qb, kb, vb, mask_x, TXc, 0, ao);
    gemm(ao, cOx + (size_t)i * Dc * Dc, box + (size_t)i * Dc, h, h, BT, Dc, Dc, Dc, 0);

    // --- FFN ---
    lnorm(h, ln_s + (size_t)(i * 3 + 2) * Dc, ln_b + (size_t)(i * 3 + 2) * Dc, y);
    gemm(y,  cW1 + (size_t)i * DFFc * Dc, b1 + (size_t)i * DFFc, nullptr, ff, BT, DFFc, Dc, DFFc, 1);
    gemm(ff, cW2 + (size_t)i * Dc * DFFc, b2 + (size_t)i * Dc,   h,       h,  BT, Dc, DFFc, Dc, 0);
  }

  // final layernorm -> y holds normalized h ("hf")
  lnorm(h, fln_s, fln_b, y);

  // vocab logits straight into d_out rows (stride OUTW), then softmax in place
  gemm(y, cProj, projb, nullptr, out, BT, VOCABc, Dc, OUTW, 0);
  softmax_vocab_kernel<<<dim3(BT), dim3(256), 0, stream>>>(out);

  // copy attention: pctx = memory @ Wc^T + b ; hW = hf @ Wcomb^T
  gemm(memory, cC,    bAtt,    nullptr, pctx, BX, Dc, Dc, Dc, 0);
  gemm(y,      cComb, nullptr, nullptr, hWb,  BT, Dc, Dc, Dc, 0);
  copy_attn_kernel<<<dim3(BT), dim3(256), 0, stream>>>(pctx, hWb, Uatt, mask_x, memory,
                                                       emb, y, vvec, bv, wbuf, gbuf);

  // gate scale + scatter-add of copy distribution
  combine_kernel<<<dim3(BT), dim3(256), 0, stream>>>(out, gbuf, wbuf, xid);

  (void)in_sizes; (void)n_in; (void)out_size; (void)ws_size;
}
